// EdgeFeatureGAT_48163763257454
// MI455X (gfx1250) — compile-verified, matrix operand order checked by reading, not run on hardware
//
#include <hip/hip_runtime.h>
#include <hip/hip_bf16.h>

#define N_NODES 50000
#define N_EDGES 400000
#define HDIM 128
#define NFEAT 10
#define EFEAT 10

typedef __attribute__((ext_vector_type(16))) __bf16        v16bf;
typedef __attribute__((ext_vector_type(8)))  float         v8f;
typedef __attribute__((ext_vector_type(8)))  unsigned int  v8u;

// ---------------------------------------------------------------------------
// bf16 pack/unpack helpers (RNE on pack; unpack is a shift)
// ---------------------------------------------------------------------------
__device__ __forceinline__ unsigned int bfbits(float f) {
  unsigned int u = __float_as_uint(f);
  u += 0x7FFFu + ((u >> 16) & 1u);
  return u >> 16;
}
__device__ __forceinline__ unsigned int pack2(float lo, float hi) {
  return bfbits(lo) | (bfbits(hi) << 16);
}
__device__ __forceinline__ float unpack_lo(unsigned int u) {
  return __uint_as_float(u << 16);
}
__device__ __forceinline__ float unpack_hi(unsigned int u) {
  return __uint_as_float(u & 0xffff0000u);
}

// ---------------------------------------------------------------------------
// async global -> LDS copy (CDNA5 GLOBAL_LOAD_ASYNC_TO_LDS_B64, ASYNCcnt)
// lds_addr: LDS byte offset (low 32 bits of generic pointer to __shared__)
// ---------------------------------------------------------------------------
__device__ __forceinline__ void async_gather_b64(unsigned int lds_addr,
                                                 const unsigned int* gptr) {
  asm volatile("global_load_async_to_lds_b64 %0, %1, off"
               :: "v"(lds_addr), "v"(gptr) : "memory");
}
__device__ __forceinline__ void wait_asynccnt0() {
  asm volatile("s_wait_asynccnt 0x0" ::: "memory");
}

// ---------------------------------------------------------------------------
// WMMA fragment loads (data already packed bf16, 2 per dword)
//
// A tile: row-major packed rows (ldu dwords per row).  ISA 16-bit A layout:
// lanes 0-15 hold K 0-7 / 16-23, lanes 16-31 hold K 8-15 / 24-31.
// Per lane: elements 0..7  -> dwords  [k0/2 + half*4 .. +3]
//           elements 8..15 -> dwords  [+8 .. +11]          (two b128 loads)
// ---------------------------------------------------------------------------
__device__ __forceinline__ v16bf load_fragA_pk(const unsigned int* base, int ldu,
                                               int m, int half, int k0) {
  const unsigned int* p = base + m * ldu + (k0 >> 1) + half * 4;
  v8u r;
#pragma unroll
  for (int j = 0; j < 4; ++j) {
    r[j]     = p[j];
    r[4 + j] = p[8 + j];
  }
  return __builtin_bit_cast(v16bf, r);
}

// B fragments are pre-swizzled into per-(ks,ct,lane) order: 8 contiguous
// dwords per lane -> two contiguous b128 loads.
__device__ __forceinline__ v16bf load_fragB_pk(const unsigned int* Wpk,
                                               int fragIdx, int lane) {
  const unsigned int* p = Wpk + ((size_t)fragIdx * 32 + lane) * 8;
  v8u r;
#pragma unroll
  for (int j = 0; j < 8; ++j) r[j] = p[j];
  return __builtin_bit_cast(v16bf, r);
}

__device__ __forceinline__ v8f wmma_bf16(v16bf a, v16bf b, v8f c) {
  return __builtin_amdgcn_wmma_f32_16x16x32_bf16(false, a, false, b,
                                                 (short)0, c, false, false);
}

__device__ __forceinline__ void atomicMaxFloat(float* addr, float value) {
  if (value >= 0.f) atomicMax((int*)addr, __float_as_int(value));
  else              atomicMin((unsigned int*)addr, __float_as_uint(value));
}

__device__ __forceinline__ float leaky02(float v) { return v > 0.f ? v : 0.2f * v; }

__device__ __forceinline__ float block_sum128(float v, float* red, int t) {
  red[t] = v;
  __syncthreads();
#pragma unroll
  for (int off = 64; off > 0; off >>= 1) {
    if (t < off) red[t] += red[t + off];
    __syncthreads();
  }
  float r = red[0];
  __syncthreads();
  return r;
}

// ---------------------------------------------------------------------------
// generic fill
// ---------------------------------------------------------------------------
__global__ void fill_k(float* __restrict__ p, float v, int n) {
  int i = blockIdx.x * blockDim.x + threadIdx.x;
  if (i < n) p[i] = v;
}

// ---------------------------------------------------------------------------
// weight pre-swizzle: W [K x ldn] fp32 -> packed bf16 fragments.
// Fragment (ks, ct), lane: dword j packs W[kb, n], W[kb+1, n] with
// kb = ks*32 + half*8 + (j<4 ? 2j : 16 + 2j - 8),  n = ct*16 + (lane&15).
// one thread = one (fragment, lane), writes 8 dwords.
// ---------------------------------------------------------------------------
__global__ void pack_weight_k(const float* __restrict__ W, unsigned int* __restrict__ out,
                              int ldn, int K, int NT) {
  const int i = blockIdx.x * blockDim.x + threadIdx.x;
  const int total = (K / 32) * NT * 32;
  if (i >= total) return;
  const int lane = i & 31;
  const int fid  = i >> 5;
  const int ct   = fid % NT;
  const int ks   = fid / NT;
  const int half = lane >> 4;
  const int n    = ct * 16 + (lane & 15);
  const int k0   = ks * 32;
  unsigned int* o = out + (size_t)i * 8;
#pragma unroll
  for (int j = 0; j < 8; ++j) {
    const int kb = k0 + half * 8 + (j < 4 ? 2 * j : 16 + 2 * j - 8);
    o[j] = pack2(W[(size_t)kb * ldn + n], W[(size_t)(kb + 1) * ldn + n]);
  }
}

// ---------------------------------------------------------------------------
// node encoder: hpk = pack_bf16(relu(LN(x @ ne_w + ne_b)))  (block=128 = 1 node)
// ---------------------------------------------------------------------------
__global__ void encode_nodes_k(const float* __restrict__ x, const float* __restrict__ w,
                               const float* __restrict__ bias, const float* __restrict__ g,
                               const float* __restrict__ b, unsigned int* __restrict__ hpk) {
  __shared__ float red[128];
  const int n = blockIdx.x, t = threadIdx.x;
  float v = bias[t];
#pragma unroll
  for (int k = 0; k < NFEAT; ++k) v += x[n * NFEAT + k] * w[k * HDIM + t];
  float s1 = block_sum128(v, red, t);
  float s2 = block_sum128(v * v, red, t);
  float mean = s1 * (1.f / HDIM);
  float inv  = rsqrtf(s2 * (1.f / HDIM) - mean * mean + 1e-5f);
  float r    = fmaxf((v - mean) * inv * g[t] + b[t], 0.f);
  float rn   = __shfl_xor(r, 1);
  if ((t & 1) == 0) hpk[(size_t)n * 64 + (t >> 1)] = pack2(r, rn);
}

// ---------------------------------------------------------------------------
// xp = h @ W  (N x 128 @ 128 x 128), one wave per 16x16 C tile.
// A from packed hpk (global b128s), B pre-swizzled.  grid=(N/16, 8), block=32.
// ---------------------------------------------------------------------------
__global__ void gemm_xw_k(const unsigned int* __restrict__ Xpk,
                          const unsigned int* __restrict__ Wpk,
                          float* __restrict__ Y) {
  const int lane = threadIdx.x;
  const int half = lane >> 4;
  const int n16  = lane & 15;
  const int rt   = blockIdx.x;
  const int ct   = blockIdx.y;
  const unsigned int* abase = Xpk + (size_t)rt * 16 * 64;
  v8f acc = {};
#pragma unroll
  for (int ks = 0; ks < 4; ++ks) {
    __builtin_prefetch(Wpk + ((size_t)((ks + 1) * 8 + ct) * 32) * 8, 0, 1);
    v16bf a = load_fragA_pk(abase, 64, n16, half, ks * 32);
    v16bf b = load_fragB_pk(Wpk, ks * 8 + ct, lane);
    acc = wmma_bf16(a, b, acc);
  }
#pragma unroll
  for (int r = 0; r < 8; ++r) {
    const int M = r + 8 * half;
    Y[(size_t)(rt * 16 + M) * HDIM + ct * 16 + n16] = acc[r];
  }
}

// ---------------------------------------------------------------------------
// attention coefficients: a_s/a_d [N, heads]
// ---------------------------------------------------------------------------
__global__ void attn_coeff_k(const float* __restrict__ xp, const float* __restrict__ asrc,
                             const float* __restrict__ adst, float* __restrict__ a_s,
                             float* __restrict__ a_d, int heads, int dout) {
  int i = blockIdx.x * blockDim.x + threadIdx.x;
  if (i >= N_NODES * heads) return;
  int node = i / heads, hd = i - (i / heads) * heads;
  float ss = 0.f, dd = 0.f;
  for (int d = 0; d < dout; ++d) {
    float v = xp[(size_t)node * HDIM + hd * dout + d];
    ss += v * asrc[hd * dout + d];
    dd += v * adst[hd * dout + d];
  }
  a_s[i] = ss;
  a_d[i] = dd;
}

// ---------------------------------------------------------------------------
// segment softmax over dst (self loops = virtual edges E..E+N)
// ---------------------------------------------------------------------------
__global__ void edge_max_k(const int* __restrict__ ei, const float* __restrict__ a_s,
                           const float* __restrict__ a_d, float* __restrict__ segm,
                           int heads) {
  int i = blockIdx.x * blockDim.x + threadIdx.x;
  int total = (N_EDGES + N_NODES) * heads;
  if (i >= total) return;
  int e = i / heads, hd = i - (i / heads) * heads;
  int s, d;
  if (e < N_EDGES) { s = ei[e]; d = ei[N_EDGES + e]; } else { s = d = e - N_EDGES; }
  float v = leaky02(a_s[s * heads + hd] + a_d[d * heads + hd]);
  atomicMaxFloat(&segm[d * heads + hd], v);
}

__global__ void edge_expsum_k(const int* __restrict__ ei, const float* __restrict__ a_s,
                              const float* __restrict__ a_d, const float* __restrict__ segm,
                              float* __restrict__ segs, float* __restrict__ exb, int heads) {
  int i = blockIdx.x * blockDim.x + threadIdx.x;
  int total = (N_EDGES + N_NODES) * heads;
  if (i >= total) return;
  int e = i / heads, hd = i - (i / heads) * heads;
  int s, d;
  if (e < N_EDGES) { s = ei[e]; d = ei[N_EDGES + e]; } else { s = d = e - N_EDGES; }
  float v  = leaky02(a_s[s * heads + hd] + a_d[d * heads + hd]);
  float ex = __expf(v - segm[d * heads + hd]);
  exb[i] = ex;
  atomicAdd(&segs[d * heads + hd], ex);
}

__global__ void edge_scatter_k(const int* __restrict__ ei, const float* __restrict__ xp,
                               const float* __restrict__ exb, const float* __restrict__ segs,
                               float* __restrict__ agg, int heads, int dout) {
  int i = blockIdx.x * blockDim.x + threadIdx.x;
  int total = (N_EDGES + N_NODES) * heads;
  if (i >= total) return;
  int e = i / heads, hd = i - (i / heads) * heads;
  int s, d;
  if (e < N_EDGES) { s = ei[e]; d = ei[N_EDGES + e]; } else { s = d = e - N_EDGES; }
  float alpha = exb[i] / segs[d * heads + hd];
  for (int dd = 0; dd < dout; ++dd)
    atomicAdd(&agg[(size_t)d * HDIM + hd * dout + dd],
              xp[(size_t)s * HDIM + hd * dout + dd] * alpha);
}

// ---------------------------------------------------------------------------
// GAT epilogue: hpk = pack_bf16(relu(LN(agg + bias)))
// ---------------------------------------------------------------------------
__global__ void ln_relu_bias_k(const float* __restrict__ in, const float* __restrict__ bias,
                               const float* __restrict__ g, const float* __restrict__ b,
                               unsigned int* __restrict__ hpk) {
  __shared__ float red[128];
  const int n = blockIdx.x, t = threadIdx.x;
  float v = in[(size_t)n * HDIM + t] + bias[t];
  float s1 = block_sum128(v, red, t);
  float s2 = block_sum128(v * v, red, t);
  float mean = s1 * (1.f / HDIM);
  float inv  = rsqrtf(s2 * (1.f / HDIM) - mean * mean + 1e-5f);
  float r    = fmaxf((v - mean) * inv * g[t] + b[t], 0.f);
  float rn   = __shfl_xor(r, 1);
  if ((t & 1) == 0) hpk[(size_t)n * 64 + (t >> 1)] = pack2(r, rn);
}

// ---------------------------------------------------------------------------
// fused edge decoder: one wave per 16-edge tile, packed-bf16 LDS tiles.
// s/d gathers use async global->LDS (ASYNCcnt) overlapped with the edge
// encoder compute.  grid = E/16, block = 32
// ---------------------------------------------------------------------------
__global__ void edge_decoder_k(
    const unsigned int* __restrict__ hpk, const int* __restrict__ edge_index,
    const float* __restrict__ edge_attr,
    const float* __restrict__ ee_w, const float* __restrict__ ee_b,
    const float* __restrict__ ee_g, const float* __restrict__ ee_beta,
    const unsigned int* __restrict__ wgate, const float* __restrict__ gate_b,
    const unsigned int* __restrict__ wc1, const float* __restrict__ c1_b,
    const float* __restrict__ cl1_g, const float* __restrict__ cl1_b,
    const unsigned int* __restrict__ wc2, const float* __restrict__ c2_b,
    const float* __restrict__ cl2_g, const float* __restrict__ cl2_b,
    const float* __restrict__ c3_w, const float* __restrict__ c3_b,
    float* __restrict__ out) {
  __shared__ unsigned int spk[16 * 64];   // s rows (later s', later z2)
  __shared__ unsigned int dpk[16 * 64];   // d rows (later d')
  __shared__ unsigned int epk[16 * 64];   // edge-encoder rows (later z1)

  const int lane = threadIdx.x;
  const int half = lane >> 4;
  const int n16  = lane & 15;
  const int eb   = blockIdx.x * 16;
  const int* row = edge_index;
  const int* col = edge_index + N_EDGES;

  // -- async gather of packed s/d node rows into LDS (no VGPR round trip) ---
  {
    const unsigned int sbase = (unsigned int)(size_t)(void*)spk;
    const unsigned int dbase = (unsigned int)(size_t)(void*)dpk;
#pragma unroll
    for (int r = 0; r < 16; ++r) {
      const unsigned int* sp = hpk + (size_t)row[eb + r] * 64 + lane * 2;
      const unsigned int* dp = hpk + (size_t)col[eb + r] * 64 + lane * 2;
      const unsigned int loff = (unsigned int)((r * 64 + lane * 2) * 4);
      async_gather_b64(sbase + loff, sp);
      async_gather_b64(dbase + loff, dp);
    }
  }

  // -- edge encoder (overlaps with async gathers): relu(LN(ea @ ee_w + b)) --
  {
    const int er = lane >> 1;     // edge row 0..15 (2 lanes / edge)
    const int part = lane & 1;    // which 64-feature half
    const size_t e = (size_t)(eb + er) * EFEAT;
    float ea[EFEAT];
#pragma unroll
    for (int k = 0; k < EFEAT; ++k) ea[k] = edge_attr[e + k];
    float s1 = 0.f, s2 = 0.f;
    for (int i = 0; i < 64; ++i) {
      const int f = part * 64 + i;
      float v = ee_b[f];
#pragma unroll
      for (int k = 0; k < EFEAT; ++k) v += ea[k] * ee_w[k * HDIM + f];
      s1 += v; s2 += v * v;
    }
    s1 += __shfl_xor(s1, 1);
    s2 += __shfl_xor(s2, 1);
    const float mean = s1 * (1.f / HDIM);
    const float inv  = rsqrtf(s2 * (1.f / HDIM) - mean * mean + 1e-5f);
    for (int i = 0; i < 32; ++i) {
      const int f = part * 64 + 2 * i;
      float v0 = ee_b[f], v1 = ee_b[f + 1];
#pragma unroll
      for (int k = 0; k < EFEAT; ++k) {
        v0 += ea[k] * ee_w[k * HDIM + f];
        v1 += ea[k] * ee_w[k * HDIM + f + 1];
      }
      v0 = fmaxf((v0 - mean) * inv * ee_g[f]     + ee_beta[f],     0.f);
      v1 = fmaxf((v1 - mean) * inv * ee_g[f + 1] + ee_beta[f + 1], 0.f);
      epk[er * 64 + part * 32 + i] = pack2(v0, v1);
    }
  }
  wait_asynccnt0();               // s/d rows resident in LDS
  __syncthreads();

  // -- gate = sigmoid([s d e] @ gate_w + b); s += g*e; d += g*e -------------
  {
    v8f acc[8];
    const v8f vz = {};
#pragma unroll
    for (int ct = 0; ct < 8; ++ct) acc[ct] = vz;
#pragma unroll
    for (int ks = 0; ks < 12; ++ks) {
      const int k0 = ks * 32;
      const unsigned int* ab = (k0 < 128) ? spk : (k0 < 256) ? dpk : epk;
      v16bf a = load_fragA_pk(ab, 64, n16, half, k0 & 127);
#pragma unroll
      for (int ct = 0; ct < 8; ++ct) {
        v16bf b = load_fragB_pk(wgate, ks * 8 + ct, lane);
        acc[ct] = wmma_bf16(a, b, acc[ct]);
      }
    }
#pragma unroll
    for (int ct = 0; ct < 8; ++ct) {
      const float gb = gate_b[ct * 16 + n16];
#pragma unroll
      for (int r = 0; r < 8; ++r) {
        const float g0 = 1.f / (1.f + __expf(-(acc[ct][r] + gb)));
        const float g1 = __shfl_xor(g0, 1);
        if ((n16 & 1) == 0) {
          const int di = (r + 8 * half) * 64 + ct * 8 + (n16 >> 1);
          const unsigned int ep = epk[di];
          const unsigned int sp = spk[di];
          const unsigned int dp = dpk[di];
          const float e0 = unpack_lo(ep), e1 = unpack_hi(ep);
          spk[di] = pack2(unpack_lo(sp) + g0 * e0, unpack_hi(sp) + g1 * e1);
          dpk[di] = pack2(unpack_lo(dp) + g0 * e0, unpack_hi(dp) + g1 * e1);
        }
      }
    }
  }
  __syncthreads();

  // -- c1: z1 = relu(LN([s' d'] @ c1_w + c1_b)) -> packed epk ---------------
  {
    v8f acc[8];
    const v8f vz = {};
#pragma unroll
    for (int ct = 0; ct < 8; ++ct) acc[ct] = vz;
#pragma unroll
    for (int ks = 0; ks < 8; ++ks) {
      const int k0 = ks * 32;
      const unsigned int* ab = (k0 < 128) ? spk : dpk;
      v16bf a = load_fragA_pk(ab, 64, n16, half, k0 & 127);
#pragma unroll
      for (int ct = 0; ct < 8; ++ct) {
        v16bf b = load_fragB_pk(wc1, ks * 8 + ct, lane);
        acc[ct] = wmma_bf16(a, b, acc[ct]);
      }
    }
#pragma unroll
    for (int ct = 0; ct < 8; ++ct) {
      const float bb = c1_b[ct * 16 + n16];
#pragma unroll
      for (int r = 0; r < 8; ++r) acc[ct][r] += bb;
    }
#pragma unroll
    for (int r = 0; r < 8; ++r) {
      float s1 = 0.f, s2 = 0.f;
#pragma unroll
      for (int ct = 0; ct < 8; ++ct) { const float v = acc[ct][r]; s1 += v; s2 += v * v; }
      for (int mask = 1; mask <= 8; mask <<= 1) {
        s1 += __shfl_xor(s1, mask);
        s2 += __shfl_xor(s2, mask);
      }
      const float mean = s1 * (1.f / 128.f);
      const float inv  = rsqrtf(s2 * (1.f / 128.f) - mean * mean + 1e-5f);
      const int M = r + 8 * half;
#pragma unroll
      for (int ct = 0; ct < 8; ++ct) {
        const int nn = ct * 16 + n16;
        const float v0 = fmaxf((acc[ct][r] - mean) * inv * cl1_g[nn] + cl1_b[nn], 0.f);
        const float v1 = __shfl_xor(v0, 1);
        if ((n16 & 1) == 0) epk[M * 64 + ct * 8 + (n16 >> 1)] = pack2(v0, v1);
      }
    }
  }
  __syncthreads();

  // -- c2: z2 = relu(LN(z1 @ c2_w + c2_b)) -> packed spk[16][32] ------------
  {
    v8f acc[4];
    const v8f vz = {};
#pragma unroll
    for (int ct = 0; ct < 4; ++ct) acc[ct] = vz;
#pragma unroll
    for (int ks = 0; ks < 4; ++ks) {
      v16bf a = load_fragA_pk(epk, 64, n16, half, ks * 32);
#pragma unroll
      for (int ct = 0; ct < 4; ++ct) {
        v16bf b = load_fragB_pk(wc2, ks * 4 + ct, lane);
        acc[ct] = wmma_bf16(a, b, acc[ct]);
      }
    }
#pragma unroll
    for (int ct = 0; ct < 4; ++ct) {
      const float bb = c2_b[ct * 16 + n16];
#pragma unroll
      for (int r = 0; r < 8; ++r) acc[ct][r] += bb;
    }
#pragma unroll
    for (int r = 0; r < 8; ++r) {
      float s1 = 0.f, s2 = 0.f;
#pragma unroll
      for (int ct = 0; ct < 4; ++ct) { const float v = acc[ct][r]; s1 += v; s2 += v * v; }
      for (int mask = 1; mask <= 8; mask <<= 1) {
        s1 += __shfl_xor(s1, mask);
        s2 += __shfl_xor(s2, mask);
      }
      const float mean = s1 * (1.f / 64.f);
      const float inv  = rsqrtf(s2 * (1.f / 64.f) - mean * mean + 1e-5f);
      const int M = r + 8 * half;
#pragma unroll
      for (int ct = 0; ct < 4; ++ct) {
        const int nn = ct * 16 + n16;
        const float v0 = fmaxf((acc[ct][r] - mean) * inv * cl2_g[nn] + cl2_b[nn], 0.f);
        const float v1 = __shfl_xor(v0, 1);
        if ((n16 & 1) == 0) spk[M * 32 + ct * 8 + (n16 >> 1)] = pack2(v0, v1);
      }
    }
  }
  __syncthreads();

  // -- c3: out = z2 @ c3_w + c3_b  (lane = (edge row, class)) ---------------
  {
    const int m = n16;
    const int cls = half;              // 2 classes across the two halves
    float v = c3_b[cls];
#pragma unroll
    for (int kd = 0; kd < 32; ++kd) {
      const unsigned int z = spk[m * 32 + kd];
      v += unpack_lo(z) * c3_w[(2 * kd) * 2 + cls]
         + unpack_hi(z) * c3_w[(2 * kd + 1) * 2 + cls];
    }
    out[(size_t)(eb + m) * 2 + cls] = v;
  }
}

// ---------------------------------------------------------------------------
// host-side pipeline
// ---------------------------------------------------------------------------
static void run_gat(hipStream_t stream, const unsigned int* hpk_in, float* agg, float* xp,
                    const unsigned int* Wpk, const float* asrc, const float* adst,
                    const float* bias, int heads, int dout,
                    const float* lng, const float* lnb,
                    float* a_s, float* a_d, float* segm, float* segs, float* exb,
                    const int* edge_index, unsigned int* hpk_out) {
  dim3 ggrid(N_NODES / 16, HDIM / 16);
  gemm_xw_k<<<ggrid, 32, 0, stream>>>(hpk_in, Wpk, xp);
  const int nh = N_NODES * heads;
  attn_coeff_k<<<(nh + 255) / 256, 256, 0, stream>>>(xp, asrc, adst, a_s, a_d, heads, dout);
  fill_k<<<(nh + 255) / 256, 256, 0, stream>>>(segm, -1e30f, nh);
  fill_k<<<(nh + 255) / 256, 256, 0, stream>>>(segs, 0.f, nh);
  fill_k<<<(N_NODES * HDIM + 255) / 256, 256, 0, stream>>>(agg, 0.f, N_NODES * HDIM);
  const int tot = (N_EDGES + N_NODES) * heads;
  edge_max_k<<<(tot + 255) / 256, 256, 0, stream>>>(edge_index, a_s, a_d, segm, heads);
  edge_expsum_k<<<(tot + 255) / 256, 256, 0, stream>>>(edge_index, a_s, a_d, segm, segs, exb, heads);
  edge_scatter_k<<<(tot + 255) / 256, 256, 0, stream>>>(edge_index, xp, exb, segs, agg, heads, dout);
  ln_relu_bias_k<<<N_NODES, 128, 0, stream>>>(agg, bias, lng, lnb, hpk_out);
}

extern "C" void kernel_launch(void* const* d_in, const int* in_sizes, int n_in,
                              void* d_out, int out_size, void* d_ws, size_t ws_size,
                              hipStream_t stream) {
  (void)in_sizes; (void)n_in; (void)out_size; (void)ws_size;
  const float* x         = (const float*)d_in[0];
  const int*   edge_index= (const int*)d_in[1];
  const float* edge_attr = (const float*)d_in[2];
  const float* ne_w      = (const float*)d_in[3];
  const float* ne_b      = (const float*)d_in[4];
  const float* ne_g      = (const float*)d_in[5];
  const float* ne_beta   = (const float*)d_in[6];
  const float* ee_w      = (const float*)d_in[7];
  const float* ee_b      = (const float*)d_in[8];
  const float* ee_g      = (const float*)d_in[9];
  const float* ee_beta   = (const float*)d_in[10];
  const float* gate_w    = (const float*)d_in[11];
  const float* gate_b    = (const float*)d_in[12];
  const float* gatA_w    = (const float*)d_in[13];
  const float* gatA_asrc = (const float*)d_in[14];
  const float* gatA_adst = (const float*)d_in[15];
  const float* gatA_bias = (const float*)d_in[16];
  const float* gatB_w    = (const float*)d_in[17];
  const float* gatB_asrc = (const float*)d_in[18];
  const float* gatB_adst = (const float*)d_in[19];
  const float* gatB_bias = (const float*)d_in[20];
  const float* ln_g      = (const float*)d_in[21];
  const float* ln_b      = (const float*)d_in[22];
  const float* c1_w      = (const float*)d_in[23];
  const float* c1_b      = (const float*)d_in[24];
  const float* cl1_g     = (const float*)d_in[25];
  const float* cl1_b     = (const float*)d_in[26];
  const float* c2_w      = (const float*)d_in[27];
  const float* c2_b      = (const float*)d_in[28];
  const float* cl2_g     = (const float*)d_in[29];
  const float* cl2_b     = (const float*)d_in[30];
  const float* c3_w      = (const float*)d_in[31];
  const float* c3_b      = (const float*)d_in[32];

  char* ws = (char*)d_ws;
  unsigned int* hpk = (unsigned int*)ws; ws += (size_t)N_NODES * 64 * 4;
  float* agg  = (float*)ws; ws += (size_t)N_NODES * HDIM * 4;
  float* xp   = (float*)ws; ws += (size_t)N_NODES * HDIM * 4;
  float* a_s  = (float*)ws; ws += (size_t)N_NODES * 8 * 4;
  float* a_d  = (float*)ws; ws += (size_t)N_NODES * 8 * 4;
  float* segm = (float*)ws; ws += (size_t)N_NODES * 8 * 4;
  float* segs = (float*)ws; ws += (size_t)N_NODES * 8 * 4;
  float* exb  = (float*)ws; ws += (size_t)(N_EDGES + N_NODES) * 8 * 4;
  unsigned int* wA0   = (unsigned int*)ws; ws += (size_t)128 * 128 * 2;  // bf16 bytes
  unsigned int* wA1   = (unsigned int*)ws; ws += (size_t)128 * 128 * 2;
  unsigned int* wB    = (unsigned int*)ws; ws += (size_t)128 * 128 * 2;
  unsigned int* wgate = (unsigned int*)ws; ws += (size_t)384 * 128 * 2;
  unsigned int* wc1   = (unsigned int*)ws; ws += (size_t)256 * 128 * 2;
  unsigned int* wc2   = (unsigned int*)ws; ws += (size_t)128 * 64 * 2;

  // ---- one-time weight pre-swizzles (cheap; run every call, deterministic)
  {
    struct PW { const float* w; unsigned int* o; int ldn; int K; };
    const PW pws[6] = {
      { gatA_w,                  wA0,   128, 128 },
      { gatA_w + 128 * 128,      wA1,   128, 128 },
      { gatB_w,                  wB,    128, 128 },
      { gate_w,                  wgate, 128, 384 },
      { c1_w,                    wc1,   128, 256 },
      { c2_w,                    wc2,    64, 128 },
    };
    for (int i = 0; i < 6; ++i) {
      const int NT = pws[i].ldn / 16;
      const int tot = (pws[i].K / 32) * NT * 32;
      pack_weight_k<<<(tot + 255) / 256, 256, 0, stream>>>(pws[i].w, pws[i].o,
                                                           pws[i].ldn, pws[i].K, NT);
    }
  }

  // ---- node encoder -> packed bf16 embeddings
  encode_nodes_k<<<N_NODES, 128, 0, stream>>>(x, ne_w, ne_b, ne_g, ne_beta, hpk);

  // ---- two 8-head GAT layers (concat) + one single-head GAT layer
  run_gat(stream, hpk, agg, xp, wA0, gatA_asrc, gatA_adst, gatA_bias, 8, 16,
          ln_g, ln_b, a_s, a_d, segm, segs, exb, edge_index, hpk);
  run_gat(stream, hpk, agg, xp, wA1, gatA_asrc + 128, gatA_adst + 128,
          gatA_bias + 128, 8, 16, ln_g + 128, ln_b + 128,
          a_s, a_d, segm, segs, exb, edge_index, hpk);
  run_gat(stream, hpk, agg, xp, wB, gatB_asrc, gatB_adst, gatB_bias, 1, HDIM,
          ln_g + 256, ln_b + 256, a_s, a_d, segm, segs, exb, edge_index, hpk);

  // ---- fused edge decoder (edge encoder + gate + c1 + c2 + c3)
  edge_decoder_k<<<N_EDGES / 16, 32, 0, stream>>>(
      hpk, edge_index, edge_attr,
      ee_w, ee_b, ee_g, ee_beta,
      wgate, gate_b,
      wc1, c1_b, cl1_g, cl1_b,
      wc2, c2_b, cl2_g, cl2_b,
      c3_w, c3_b,
      (float*)d_out);
}
